// SymmetricContraction_41102837023172
// MI455X (gfx1250) — matrix operand order repaired and checked by reading, not compiled, not run on hardware
//
#include <hip/hip_runtime.h>

typedef __attribute__((ext_vector_type(16))) __bf16   v16bf;
typedef __attribute__((ext_vector_type(8)))  float    v8f;
typedef __attribute__((ext_vector_type(4)))  unsigned u32x4;
typedef __attribute__((ext_vector_type(8)))  unsigned u32x8;

#define C_CH   128
#define E_ELEM 10

__device__ __forceinline__ void tdm_wait0() {
#if __has_builtin(__builtin_amdgcn_s_wait_tensorcnt)
  __builtin_amdgcn_s_wait_tensorcnt(0);
#else
  asm volatile("s_wait_tensorcnt 0x0" ::: "memory");
#endif
}

// Issue one TDM 1-D tile copy: nelem bf16 elements, contiguous, global -> LDS.
// D# group0: count=1, lds_addr, global_addr(57b), type=2.
// D# group1: data_size=1 (2B), tensor_dim0 huge (no OOB clipping), tile_dim0=nelem.
__device__ __forceinline__ void tdm_load_panel(void* lds_dst, const void* gsrc,
                                               unsigned nelem) {
  unsigned lds_off = (unsigned)(unsigned long long)lds_dst;  // low 32b = LDS offset
  unsigned long long ga = (unsigned long long)gsrc;
  unsigned td0 = 0x40000000u;  // tensor_dim0: large so no OOB zeroing
  u32x4 g0;
  g0[0] = (unsigned)__builtin_amdgcn_readfirstlane((int)1u);
  g0[1] = (unsigned)__builtin_amdgcn_readfirstlane((int)lds_off);
  g0[2] = (unsigned)__builtin_amdgcn_readfirstlane((int)(unsigned)(ga & 0xffffffffull));
  g0[3] = (unsigned)__builtin_amdgcn_readfirstlane(
              (int)((((unsigned)(ga >> 32)) & 0x01ffffffu) | 0x80000000u));
  u32x8 g1;
  g1[0] = (unsigned)__builtin_amdgcn_readfirstlane((int)0x00010000u);        // data_size=2B
  g1[1] = (unsigned)__builtin_amdgcn_readfirstlane((int)((td0 & 0xffffu) << 16));
  g1[2] = (unsigned)__builtin_amdgcn_readfirstlane((int)((td0 >> 16) | (1u << 16)));
  g1[3] = (unsigned)__builtin_amdgcn_readfirstlane((int)((nelem & 0xffffu) << 16)); // tile_dim0
  g1[4] = (unsigned)__builtin_amdgcn_readfirstlane((int)1u);                 // tile_dim1=1
  g1[5] = (unsigned)__builtin_amdgcn_readfirstlane((int)td0);                // dim0 stride
  g1[6] = (unsigned)__builtin_amdgcn_readfirstlane((int)0u);
  g1[7] = (unsigned)__builtin_amdgcn_readfirstlane((int)0u);
  asm volatile("tensor_load_to_lds %0, %1" :: "s"(g0), "s"(g1) : "memory");
}

// ---------------------------------------------------------------------------
// Prep: panelized bf16 B matrix. Panel p holds K rows kk = 32p..32p+31, each
// row padded to NP = N+8 elements (bank-conflict pad). Element (kk,n):
//   Bm[p*32*NP + (kk&31)*NP + n]
// Row kk = k*16+i of U3 (k<K3) then K2 rows of U2, zero padding elsewhere.
// ---------------------------------------------------------------------------
__global__ void build_bmat(const float* __restrict__ U3, const float* __restrict__ U2,
                           __bf16* __restrict__ Bm, int M, int K3, int K2, int KPAD) {
  int N = M * 256, NP = N + 8;
  int total = KPAD * NP;
  int idx = blockIdx.x * 256 + threadIdx.x;
  if (idx >= total) return;
  int p = idx / (32 * NP), rem = idx % (32 * NP);
  int kl = rem / NP, n = rem % NP;
  int kk = p * 32 + kl;
  int KI = 16 * K3;
  float v = 0.f;
  if (n < N) {
    int m = n >> 8, j = (n >> 4) & 15, l = n & 15;
    if (kk < KI) {
      int k = kk >> 4, i = kk & 15;
      v = U3[((((m * 16 + i) * 16 + j) * 16 + l) * K3) + k];
    } else if (kk < KI + K2) {
      v = U2[(((m * 16 + j) * 16 + l) * K2) + (kk - KI)];
    }
  }
  Bm[idx] = (__bf16)v;
}

// ---------------------------------------------------------------------------
// Main fused contraction. Block = 1024 thr = 32 waves = 8 row-tiles x 4 column
// groups; covers one full node (all 128 channels). Each TDM-staged K panel is
// reused by 8 row-tile groups (8x L2 traffic reduction vs per-tile loads).
// ---------------------------------------------------------------------------
template <int MOUT, int KPAD>
__global__ __launch_bounds__(1024) void mace_contract(
    const float* __restrict__ x, const float* __restrict__ y,
    const __bf16* __restrict__ Bmat,
    const float* __restrict__ W3, const float* __restrict__ W2,
    const float* __restrict__ W1, const float* __restrict__ U1,
    int K3, int K2, int K1,
    float* __restrict__ out, int out_col_ofs) {
  constexpr int N      = MOUT * 256;
  constexpr int NP     = N + 8;        // padded row length (elements)
  constexpr int NT     = N / 16;       // column tiles
  constexpr int NTPW   = NT / 4;       // per wave (4 or 12)
  constexpr int KSTEPS = KPAD / 32;
  constexpr int KPADP  = KPAD + 8;     // padded Z row
  constexpr int PANEL  = 32 * NP;      // elements per K panel

  extern __shared__ __align__(16) char smem[];
  __bf16* Bb = (__bf16*)smem;                                   // [2][PANEL]
  __bf16* Zs = (__bf16*)(smem + 4 * PANEL);                     // [128][KPADP]
  float*  Xl = (float*)(smem + 4 * PANEL + 256 * KPADP);        // [128][16]
  float*  B2 = Xl + 128 * 16;                                   // [128][MOUT*16]

  const int tid  = threadIdx.x;
  const int wave = tid >> 5, lane = tid & 31;
  const int rtw  = wave >> 2;          // row-tile group (0..7)
  const int cg   = wave & 3;           // column group
  const int b    = blockIdx.x;         // one node per block

  int e = 0;
  for (int t = 0; t < E_ELEM; ++t)
    if (y[b * E_ELEM + t] > 0.5f) e = t;

  for (int idx = tid; idx < 128 * 16; idx += 1024) {
    Xl[idx] = x[(long)b * C_CH * 16 + idx];
  }
  __syncthreads();

  // Z[r][k*16+i] = W3[e,k,c]*x ; K2 cols of W2 ; zero pad (incl. row pad)
  const int KI = 16 * K3;
  for (int idx = tid; idx < 128 * KPADP; idx += 1024) {
    int r = idx / KPADP, kk = idx % KPADP;
    float v = 0.f;
    if (kk < KI) {
      int k = kk >> 4, i = kk & 15;
      v = W3[(e * K3 + k) * C_CH + r] * Xl[r * 16 + i];
    } else if (kk < KI + K2) {
      v = W2[(e * K2 + (kk - KI)) * C_CH + r];
    }
    Zs[idx] = (__bf16)v;
  }

  if (wave == 0) tdm_load_panel(Bb, Bmat, PANEL);  // prefetch panel 0
  __syncthreads();                                 // Z ready

  v8f acc[NTPW];
#pragma unroll
  for (int t = 0; t < NTPW; ++t)
#pragma unroll
    for (int q = 0; q < 8; ++q) acc[t][q] = 0.f;

  const int hf = lane >> 4;            // lane half -> K sub-block (ISA A layout)
  const int rA = lane & 15;            // A row within row tile
  const __bf16* Zr = Zs + (rtw * 16 + rA) * KPADP;

  int cur = 0;
  for (int ks = 0; ks < KSTEPS; ++ks) {
    if (wave == 0) tdm_wait0();        // panel[cur] landed
    __syncthreads();                   // ...and prev compute done everywhere
    if (wave == 0 && ks + 1 < KSTEPS)
      tdm_load_panel(Bb + (cur ^ 1) * PANEL,
                     Bmat + (long)(ks + 1) * PANEL, PANEL);  // overlap with wmma

    const int k0 = ks * 32;
    v16bf a;                           // A frag: V0-3 K=+0..7, V4-7 K=+16..23 (hf +8)
#pragma unroll
    for (int v = 0; v < 4; ++v) {
      int kk = k0 + hf * 8 + 2 * v;
      a[2 * v]     = Zr[kk];
      a[2 * v + 1] = Zr[kk + 1];
      int kk2 = k0 + 16 + hf * 8 + 2 * v;
      a[2 * (v + 4)]     = Zr[kk2];
      a[2 * (v + 4) + 1] = Zr[kk2 + 1];
    }
    const __bf16* bp = Bb + cur * PANEL + (lane & 15);  // B frag col = lane&15 (N)
    auto loadB = [&](int t) {
      v16bf r;
      const int n0 = (cg * NTPW + t) * 16;
#pragma unroll
      for (int v = 0; v < 8; ++v) {
        int kb = (v < 4) ? (hf * 8 + 2 * v) : (16 + hf * 8 + 2 * (v - 4));
        r[2 * v]     = bp[kb * NP + n0];
        r[2 * v + 1] = bp[(kb + 1) * NP + n0];
      }
      return r;
    };
    v16bf bb = loadB(0);               // register double-buffer: prefetch t+1,
#pragma unroll
    for (int t = 0; t < NTPW; ++t) {   // wmma on t (partial dscnt waits)
      v16bf nx = bb;
      if (t + 1 < NTPW) nx = loadB(t + 1);
      acc[t] = __builtin_amdgcn_wmma_f32_16x16x32_bf16(
          false, a, false, bb, (short)0, acc[t], false, false);
      bb = nx;
    }
    cur ^= 1;
  }

  // B2[r][m,j] = sum_l D[r][l]*x[r][l]  (C/D layout: N index == l == lane&15)
  const int l = lane & 15;
#pragma unroll
  for (int t = 0; t < NTPW; ++t) {
    const int mj = cg * NTPW + t;      // tile index == m*16+j
#pragma unroll
    for (int v = 0; v < 8; ++v) {
      int r = rtw * 16 + v + 8 * hf;
      float p = acc[t][v] * Xl[r * 16 + l];
      p += __shfl_xor(p, 1, 32);
      p += __shfl_xor(p, 2, 32);
      p += __shfl_xor(p, 4, 32);
      p += __shfl_xor(p, 8, 32);
      if (l == 0) B2[r * (MOUT * 16) + mj] = p;
    }
  }
  __syncthreads();

  // out[b, ofs + c*MOUT + m] = sum_j (B2 + U1*W1) * x[r][j]
  if (tid < 128 * MOUT) {
    int r = tid & 127, m = tid >> 7;
    float s = 0.f;
    for (int j = 0; j < 16; ++j) {
      float b2 = B2[r * (MOUT * 16) + m * 16 + j];
      for (int k1 = 0; k1 < K1; ++k1)
        b2 += U1[(m * 16 + j) * K1 + k1] * W1[(e * K1 + k1) * C_CH + r];
      s += b2 * Xl[r * 16 + j];
    }
    out[(long)b * (4 * C_CH) + out_col_ofs + r * MOUT + m] = s;
  }
}

extern "C" void kernel_launch(void* const* d_in, const int* in_sizes, int n_in,
                              void* d_out, int out_size, void* d_ws, size_t ws_size,
                              hipStream_t stream) {
  const float* x    = (const float*)d_in[0];
  const float* y    = (const float*)d_in[1];
  const float* U3_0 = (const float*)d_in[2];
  const float* U2_0 = (const float*)d_in[3];
  const float* U1_0 = (const float*)d_in[4];
  const float* W3_0 = (const float*)d_in[5];
  const float* W2_0 = (const float*)d_in[6];
  const float* W1_0 = (const float*)d_in[7];
  const float* U3_1 = (const float*)d_in[8];
  const float* U2_1 = (const float*)d_in[9];
  const float* U1_1 = (const float*)d_in[10];
  const float* W3_1 = (const float*)d_in[11];
  const float* W2_1 = (const float*)d_in[12];
  const float* W1_1 = (const float*)d_in[13];
  float* out = (float*)d_out;

  // ws: Bmat0 bf16[384*264] then Bmat1 bf16[544*776]  (~1.05 MB)
  __bf16* Bm0 = (__bf16*)d_ws;
  __bf16* Bm1 = (__bf16*)((char*)d_ws + 384 * 264 * sizeof(__bf16));

  {
    int tot0 = 384 * 264;
    build_bmat<<<(tot0 + 255) / 256, 256, 0, stream>>>(U3_0, U2_0, Bm0, 1, 23, 4, 384);
    int tot1 = 544 * 776;
    build_bmat<<<(tot1 + 255) / 256, 256, 0, stream>>>(U3_1, U2_1, Bm1, 3, 33, 5, 544);
  }

  dim3 grid(2048);  // one node per block (all 128 channels)
  // dynamic LDS: 2 panels + padded Z + x + b2
  size_t sm0 = 4u * (32 * 264) + 256u * 392 + 8192u + 128u * 16 * 4;   // 150528
  size_t sm1 = 4u * (32 * 776) + 256u * 552 + 8192u + 128u * 48 * 4;   // 273408
  mace_contract<1, 384><<<grid, 1024, sm0, stream>>>(x, y, Bm0, W3_0, W2_0, W1_0, U1_0,
                                                     23, 4, 1, out, 0);
  mace_contract<3, 544><<<grid, 1024, sm1, stream>>>(x, y, Bm1, W3_1, W2_1, W1_1, U1_1,
                                                     33, 5, 1, out, 128);
}
